// NeuralNetwork_69990787056184
// MI455X (gfx1250) — matrix operand order repaired and checked by reading, not compiled, not run on hardware
//
#include <hip/hip_runtime.h>

// ---------------------------------------------------------------------------
// 100 x (h = h @ W^T + b), M=16384, N=K=512, f32 reference.
// bf16x3 split-precision GEMM on V_WMMA_F32_16X16X32_BF16:
//   x = x_hi + x_lo, w = w_hi + w_lo;  acc += hi*hi + hi*lo + lo*hi
// Activations stay split in ws between layers; weights split once per call.
// Staging: GLOBAL_LOAD_ASYNC_TO_LDS_B128 (ASYNCcnt) when toolchain exposes
// the builtin, else global->VGPR->LDS fallback.
// ---------------------------------------------------------------------------

typedef __attribute__((ext_vector_type(16))) __bf16 v16bf;
typedef __attribute__((ext_vector_type(4)))  __bf16 v4bf;
typedef __attribute__((ext_vector_type(8)))  float  v8f;
typedef __attribute__((ext_vector_type(4)))  float  v4f;
typedef __attribute__((ext_vector_type(4)))  unsigned int v4u;

// exact pointee type the async-LDS builtin wants: int4
typedef int i32x4 __attribute__((vector_size(16)));
typedef __attribute__((address_space(1))) i32x4 gas_i32x4;   // global (AS1)
typedef __attribute__((address_space(3))) i32x4 lds_i32x4;   // LDS    (AS3)

union Frag { v16bf v; v4u q[2]; };

#define MDIM 16384
#define NDIM 512
#define KDIM 512
#define LS   40              // LDS row stride in bf16 elems (80 B: 16B-aligned frags)
#define TI   (128 * LS)      // one 128x32 tile in LDS (elems)

#if defined(__AMDGCN__) && __has_builtin(__builtin_amdgcn_global_load_async_to_lds_b128)
#define USE_ASYNC 1
#else
#define USE_ASYNC 0
#endif

#if USE_ASYNC
__device__ __forceinline__ void async_wait0() {
#if __has_builtin(__builtin_amdgcn_s_wait_asynccnt)
    __builtin_amdgcn_s_wait_asynccnt(0);
#else
    asm volatile("s_wait_asynccnt 0" ::: "memory");
#endif
}
__device__ __forceinline__ void async_b128(const void* g, const void* l) {
    __builtin_amdgcn_global_load_async_to_lds_b128(
        (gas_i32x4*)(unsigned long long)(uintptr_t)g,
        (lds_i32x4*)(unsigned long long)(uintptr_t)l, 0, 0);
}
#endif

// ---- elementwise f32 -> (bf16 hi, bf16 lo) splitter -----------------------
__global__ __launch_bounds__(256)
void k_split(const float* __restrict__ in, __bf16* __restrict__ hi,
             __bf16* __restrict__ lo, int n4)
{
    int i = blockIdx.x * 256 + threadIdx.x;
    if (i >= n4) return;
    v4f v = ((const v4f*)in)[i];
    v4bf h, w;
#pragma unroll
    for (int j = 0; j < 4; ++j) {
        __bf16 hh = (__bf16)v[j];
        h[j] = hh;
        w[j] = (__bf16)(v[j] - (float)hh);
    }
    ((v4bf*)hi)[i] = h;
    ((v4bf*)lo)[i] = w;
}

// ---- bf16x3 GEMM: Y[m,n] = sum_k X[m,k] * W[n,k] + bias[n] ----------------
// Grid: (N/128, M/128), block: 256 threads (8 waves; 4(M) x 2(N) wave grid).
// Each wave: 2x4 grid of 16x16 WMMA accumulators (32 x 64 output).
template<int WF>
__global__ __launch_bounds__(256)
void k_gemm(const __bf16* __restrict__ Xhi, const __bf16* __restrict__ Xlo,
            const __bf16* __restrict__ Whi, const __bf16* __restrict__ Wlo,
            const float*  __restrict__ bias,
            __bf16* __restrict__ Yhi, __bf16* __restrict__ Ylo,
            float* __restrict__ Yf)
{
    __shared__ __bf16 smem[2 * 4 * TI];   // double-buffered {Ahi,Alo,Whi,Wlo}

    const int tid  = threadIdx.x;
    const int lane = tid & 31;
    const int wid  = tid >> 5;
    const int l    = lane & 15;           // column/row within 16-wide frag
    const int hf   = lane >> 4;           // half-wave select
    const int wm   = wid >> 1;            // 0..3
    const int wn   = wid & 1;             // 0..1
    const int mo   = blockIdx.y * 128;
    const int no   = blockIdx.x * 128;

    // ---- global staging: 128x32 bf16 tile = 512 x 16B units; 2 units/thread
    const int r0 = tid >> 2;              // 0..63
    const int c0 = (tid & 3) * 8;         // bf16 col 0/8/16/24

    const __bf16* gXh0 = Xhi + (size_t)(mo + r0)      * KDIM + c0;
    const __bf16* gXh1 = Xhi + (size_t)(mo + r0 + 64) * KDIM + c0;
    const __bf16* gXl0 = Xlo + (size_t)(mo + r0)      * KDIM + c0;
    const __bf16* gXl1 = Xlo + (size_t)(mo + r0 + 64) * KDIM + c0;
    const __bf16* gWh0 = Whi + (size_t)(no + r0)      * KDIM + c0;
    const __bf16* gWh1 = Whi + (size_t)(no + r0 + 64) * KDIM + c0;
    const __bf16* gWl0 = Wlo + (size_t)(no + r0)      * KDIM + c0;
    const __bf16* gWl1 = Wlo + (size_t)(no + r0 + 64) * KDIM + c0;

    v8f acc[8] = {};

#if USE_ASYNC
    const int so0 = ( r0       * LS + c0) * 2;   // LDS byte offsets of units
    const int so1 = ((r0 + 64) * LS + c0) * 2;
    auto aload = [&](int kb, int buf) {
        char* sb = (char*)smem + (size_t)buf * (4 * TI * 2);
        async_b128(gXh0 + kb, sb + 0*TI*2 + so0);
        async_b128(gXh1 + kb, sb + 0*TI*2 + so1);
        async_b128(gXl0 + kb, sb + 1*TI*2 + so0);
        async_b128(gXl1 + kb, sb + 1*TI*2 + so1);
        async_b128(gWh0 + kb, sb + 2*TI*2 + so0);
        async_b128(gWh1 + kb, sb + 2*TI*2 + so1);
        async_b128(gWl0 + kb, sb + 3*TI*2 + so0);
        async_b128(gWl1 + kb, sb + 3*TI*2 + so1);
    };
#else
    auto gload = [&](int kb, v4u* p) {
        p[0] = *(const v4u*)(gXh0 + kb);
        p[1] = *(const v4u*)(gXh1 + kb);
        p[2] = *(const v4u*)(gXl0 + kb);
        p[3] = *(const v4u*)(gXl1 + kb);
        p[4] = *(const v4u*)(gWh0 + kb);
        p[5] = *(const v4u*)(gWh1 + kb);
        p[6] = *(const v4u*)(gWl0 + kb);
        p[7] = *(const v4u*)(gWl1 + kb);
    };
    auto sstore = [&](int buf, const v4u* p) {
        __bf16* sb = smem + buf * 4 * TI;
        *(v4u*)(sb + 0*TI +  r0      * LS + c0) = p[0];
        *(v4u*)(sb + 0*TI + (r0+64)  * LS + c0) = p[1];
        *(v4u*)(sb + 1*TI +  r0      * LS + c0) = p[2];
        *(v4u*)(sb + 1*TI + (r0+64)  * LS + c0) = p[3];
        *(v4u*)(sb + 2*TI +  r0      * LS + c0) = p[4];
        *(v4u*)(sb + 2*TI + (r0+64)  * LS + c0) = p[5];
        *(v4u*)(sb + 3*TI +  r0      * LS + c0) = p[6];
        *(v4u*)(sb + 3*TI + (r0+64)  * LS + c0) = p[7];
    };
#endif

    auto compute = [&](int buf) {
        const __bf16* sb = smem + buf * 4 * TI;
        // A frags (16x32 bf16): q0 = K[8h..8h+7], q1 = K[16+8h..16+8h+7]
        Frag ah[2], al[2];
#pragma unroll
        for (int am = 0; am < 2; ++am) {
            int m = wm * 32 + am * 16 + l;
            const __bf16* pa = sb + 0*TI + m * LS + 8 * hf;
            ah[am].q[0] = *(const v4u*)(pa);
            ah[am].q[1] = *(const v4u*)(pa + 16);
            const __bf16* pl = sb + 1*TI + m * LS + 8 * hf;
            al[am].q[0] = *(const v4u*)(pl);
            al[am].q[1] = *(const v4u*)(pl + 16);
        }
        // B frags (32x16 bf16): lane holds col n=l, q0 = K[8h..8h+7], q1 = K[16+8h..16+8h+7]
        Frag bh[4], bl[4];
#pragma unroll
        for (int bn = 0; bn < 4; ++bn) {
            int n = wn * 64 + bn * 16 + l;
            const __bf16* pb = sb + 2*TI + n * LS + 8 * hf;
            bh[bn].q[0] = *(const v4u*)(pb);
            bh[bn].q[1] = *(const v4u*)(pb + 16);
            const __bf16* pq = sb + 3*TI + n * LS + 8 * hf;
            bl[bn].q[0] = *(const v4u*)(pq);
            bl[bn].q[1] = *(const v4u*)(pq + 16);
        }
#pragma unroll
        for (int am = 0; am < 2; ++am)
#pragma unroll
            for (int bn = 0; bn < 4; ++bn) {
                v8f c = acc[am * 4 + bn];
                c = __builtin_amdgcn_wmma_f32_16x16x32_bf16(
                        false, ah[am].v, false, bh[bn].v, (short)0, c, false, false);
                c = __builtin_amdgcn_wmma_f32_16x16x32_bf16(
                        false, ah[am].v, false, bl[bn].v, (short)0, c, false, false);
                c = __builtin_amdgcn_wmma_f32_16x16x32_bf16(
                        false, al[am].v, false, bh[bn].v, (short)0, c, false, false);
                acc[am * 4 + bn] = c;
            }
    };

    // ---- software pipeline over K: 16 chunks of 32, double-buffered LDS ----
#if USE_ASYNC
    aload(0, 0);
    async_wait0();
    __syncthreads();
    for (int t = 0; t < 16; ++t) {
        if (t < 15) aload((t + 1) * 32, (t + 1) & 1);   // into other buffer
        compute(t & 1);
        if (t < 15) async_wait0();                       // next buffer landed
        __syncthreads();
    }
#else
    v4u pf[8];
    gload(0, pf);
    sstore(0, pf);
    __syncthreads();
    for (int t = 0; t < 16; ++t) {
        if (t < 15) gload((t + 1) * 32, pf);
        compute(t & 1);
        if (t < 15) sstore((t + 1) & 1, pf);   // other buffer: safe pre-barrier
        __syncthreads();
    }
#endif

    // ---- epilogue: bias add; write split bf16 (or f32 for final layer) ----
#pragma unroll
    for (int bn = 0; bn < 4; ++bn) {
        int n = no + wn * 64 + bn * 16 + l;
        float bv = bias[n];
#pragma unroll
        for (int am = 0; am < 2; ++am) {
            v8f c = acc[am * 4 + bn];
            int mbase = mo + wm * 32 + am * 16 + 8 * hf;
#pragma unroll
            for (int r = 0; r < 8; ++r) {
                float y = c[r] + bv;
                size_t idx = (size_t)(mbase + r) * NDIM + n;
                if (WF) {
                    Yf[idx] = y;
                } else {
                    __bf16 hh = (__bf16)y;
                    Yhi[idx] = hh;
                    Ylo[idx] = (__bf16)(y - (float)hh);
                }
            }
        }
    }
}

// ---------------------------------------------------------------------------
extern "C" void kernel_launch(void* const* d_in, const int* in_sizes, int n_in,
                              void* d_out, int out_size, void* d_ws, size_t ws_size,
                              hipStream_t stream)
{
    const float* x  = (const float*)d_in[0];
    const float* Ws = (const float*)d_in[1];
    const float* bs = (const float*)d_in[2];
    float* out = (float*)d_out;

    const size_t XE = (size_t)MDIM * KDIM;   // 8.39M elems
    const size_t WE = (size_t)NDIM * KDIM;   // 262144 elems

    __bf16* ws = (__bf16*)d_ws;
    __bf16* xh[2] = { ws,           ws + 2 * XE };
    __bf16* xl[2] = { ws + XE,      ws + 3 * XE };
    __bf16* warea = ws + 4 * XE;

    // big path: all 100 layer weights pre-split once (~172 MB total ws)
    const bool big = ws_size >= (4 * XE + 200 * WE) * sizeof(unsigned short);

    // split input activations once
    k_split<<<(int)(XE / 4 / 256), 256, 0, stream>>>(x, xh[0], xl[0], (int)(XE / 4));
    if (big) {
        k_split<<<(int)(100 * WE / 4 / 256), 256, 0, stream>>>(
            Ws, warea, warea + 100 * WE, (int)(100 * WE / 4));
    }

    int cur = 0;
    for (int lyr = 0; lyr < 100; ++lyr) {
        const __bf16 *wh, *wl;
        if (big) {
            wh = warea + (size_t)lyr * WE;
            wl = warea + 100 * WE + (size_t)lyr * WE;
        } else {
            k_split<<<(int)(WE / 4 / 256), 256, 0, stream>>>(
                Ws + (size_t)lyr * WE, warea, warea + WE, (int)(WE / 4));
            wh = warea;
            wl = warea + WE;
        }
        int nxt = cur ^ 1;
        if (lyr == 99) {
            k_gemm<1><<<dim3(NDIM / 128, MDIM / 128), 256, 0, stream>>>(
                xh[cur], xl[cur], wh, wl, bs + (size_t)lyr * NDIM,
                xh[nxt], xl[nxt], out);
        } else {
            k_gemm<0><<<dim3(NDIM / 128, MDIM / 128), 256, 0, stream>>>(
                xh[cur], xl[cur], wh, wl, bs + (size_t)lyr * NDIM,
                xh[nxt], xl[nxt], out);
        }
        cur = nxt;
    }
}
